// RoPEAttention_22385369546993
// MI455X (gfx1250) — matrix operand order repaired and checked by reading, not compiled, not run on hardware
//
#include <hip/hip_runtime.h>

typedef __bf16 bf16;
typedef __attribute__((ext_vector_type(16))) __bf16 v16bf;
typedef __attribute__((ext_vector_type(8)))  __bf16 v8bf;
typedef __attribute__((ext_vector_type(4)))  __bf16 v4bf;
typedef __attribute__((ext_vector_type(8)))  float   v8f;

#define DEV __device__ __forceinline__

static constexpr int Bn   = 4;
static constexpr int Nseq = 2048;
static constexpr int Cch  = 1024;
static constexpr int Hh   = 16;
static constexpr int Dh   = 64;
static constexpr int ROWS = Bn * Nseq;           // 8192 flattened tokens
static constexpr float QK_SCALE = 0.125f;        // D^-0.5

// attention tiling
static constexpr int QW  = 32;                   // queries per wave (2 WMMA M-tiles)
static constexpr int WVS = 4;                    // waves per block
static constexpr int QB  = QW * WVS;             // 128 queries per block
static constexpr int KB  = 32;                   // keys per inner step

union BF16x16 { v16bf v; v8bf h[2]; };

DEV v8f wmma_bf16(v16bf a, v16bf b, v8f c) {
  // v_wmma_f32_16x16x32_bf16  D = A(16x32) * B(32x16) + C(16x16 f32)
  return __builtin_amdgcn_wmma_f32_16x16x32_bf16(
      false, a, false, b, (short)0, c, false, false);
}

// A operand (16x32 bf16, ISA 7.12.2): lanes 0-15: row M=lane, K={0..7,16..23};
// lanes 16-31: row M=lane-16, K={8..15,24..31}. row_base points at (row, k0).
DEV v16bf load_a(const bf16* row_base, int lane) {
  const int half = (lane >> 4) & 1;
  BF16x16 u;
  u.h[0] = *reinterpret_cast<const v8bf*>(row_base + half * 8);
  u.h[1] = *reinterpret_cast<const v8bf*>(row_base + half * 8 + 16);
  return u.v;
}

// B operand (32x16 bf16): lane holds column N=lane%16; lanes 0-15 K={0..7,16..23},
// lanes 16-31 K={8..15,24..31}. col_base points at column's (k0) element.
DEV v16bf load_b(const bf16* col_base, int lane) {
  const int half = (lane >> 4) & 1;
  BF16x16 u;
  u.h[0] = *reinterpret_cast<const v8bf*>(col_base + half * 8);
  u.h[1] = *reinterpret_cast<const v8bf*>(col_base + half * 8 + 16);
  return u.v;
}

// ---------------------------------------------------------------- cvt f32->bf16
__global__ void __launch_bounds__(256) cvt_ker(const float* __restrict__ in,
                                               bf16* __restrict__ out, int n) {
  int i = (blockIdx.x * blockDim.x + threadIdx.x) * 4;
  if (i + 3 < n) {
    float4 f = *reinterpret_cast<const float4*>(in + i);
    v4bf o;
    o[0] = (bf16)f.x; o[1] = (bf16)f.y; o[2] = (bf16)f.z; o[3] = (bf16)f.w;
    *reinterpret_cast<v4bf*>(out + i) = o;
  }
}

// ---------------------------------------------------------------- QKV + RoPE
// out = x @ Wqkv^T ; wave tile M=32, N=64 (one head of one of q/k/v).
// q,k get RoPE (pair partner = adjacent lane); q scaled; v stored transposed.
__global__ void __launch_bounds__(256) qkv_rope_ker(
    const bf16* __restrict__ xb, const bf16* __restrict__ wb,
    const float* __restrict__ fcos, const float* __restrict__ fsin,
    bf16* __restrict__ qo, bf16* __restrict__ ko, bf16* __restrict__ vT) {
  const int lane  = threadIdx.x & 31;
  const int lr    = lane & 15;
  const int half  = (lane >> 4) & 1;
  const int wid   = blockIdx.x * (blockDim.x >> 5) + (threadIdx.x >> 5);
  const int NT    = (3 * Cch) / 64;               // 48 N-tiles
  const int mbase = (wid / NT) * 32;
  const int nbase = (wid % NT) * 64;

  v8f c[2][4];
  #pragma unroll
  for (int a = 0; a < 2; ++a)
    #pragma unroll
    for (int t = 0; t < 4; ++t)
      #pragma unroll
      for (int i = 0; i < 8; ++i) c[a][t][i] = 0.f;

  for (int kk = 0; kk < Cch; kk += 32) {
    v16bf a0 = load_a(xb + (size_t)(mbase + lr) * Cch + kk, lane);
    v16bf a1 = load_a(xb + (size_t)(mbase + 16 + lr) * Cch + kk, lane);
    #pragma unroll
    for (int t = 0; t < 4; ++t) {
      v16bf bop = load_b(wb + (size_t)(nbase + t * 16 + lr) * Cch + kk, lane);
      c[0][t] = wmma_bf16(a0, bop, c[0][t]);
      c[1][t] = wmma_bf16(a1, bop, c[1][t]);
    }
  }

  const int which = nbase / Cch;                  // 0=q 1=k 2=v
  const int h     = (nbase % Cch) / Dh;

  #pragma unroll
  for (int a = 0; a < 2; ++a) {
    #pragma unroll
    for (int t = 0; t < 4; ++t) {
      #pragma unroll
      for (int i = 0; i < 8; ++i) {
        const int tok = mbase + a * 16 + i + half * 8;   // b*N + n
        const int d   = t * 16 + lr;
        const int b   = tok >> 11;
        const int n   = tok & (Nseq - 1);
        float f = c[a][t][i];
        if (which < 2) {
          // RoPE: pairs (2i,2i+1) sit on adjacent lanes (d parity == lane parity)
          const float partner = __shfl_xor(f, 1, 32);
          const float cs = fcos[(size_t)tok * (Dh / 2) + (d >> 1)];
          const float sn = fsin[(size_t)tok * (Dh / 2) + (d >> 1)];
          float r = (d & 1) ? (partner * sn + f * cs)    // t0*sin + t1*cos
                            : (f * cs - partner * sn);   // t0*cos - t1*sin
          if (which == 0) {
            r *= QK_SCALE;
            qo[((size_t)(b * Hh + h) * Nseq + n) * Dh + d] = (bf16)r;
          } else {
            ko[((size_t)(b * Hh + h) * Nseq + n) * Dh + d] = (bf16)r;
          }
        } else {
          // v stored transposed [B,H,D,N] so P*V B-operand loads are contiguous
          vT[((size_t)(b * Hh + h) * Dh + d) * Nseq + n] = (bf16)f;
        }
      }
    }
  }
}

// ---------------------------------------------------------------- attention
// Block = 4 waves x 32 queries on one (b,h); waves share each 32-key K/V tile
// through LDS (4x less K/V L2 traffic than per-wave streaming). Streaming
// softmax; S and P*V on WMMA; P transposed C-layout -> A-layout via LDS.
__global__ void __launch_bounds__(128) attn_ker(
    const bf16* __restrict__ q, const bf16* __restrict__ k,
    const bf16* __restrict__ vT, const bf16* __restrict__ maskb,
    bf16* __restrict__ ao) {
  __shared__ __align__(16) bf16 kls[KB * Dh];        // K tile [key][d]   4KB
  __shared__ __align__(16) bf16 vls[Dh * KB];        // V tile [d][key]   4KB
  __shared__ __align__(16) bf16 pls[WVS][2 * 16 * 32]; // per-wave P      8KB

  const int tid  = threadIdx.x;
  const int wv   = tid >> 5;
  const int lane = tid & 31;
  const int lr   = lane & 15;
  const int half = (lane >> 4) & 1;

  const int strips = Nseq / QB;                     // 16
  const int bh = blockIdx.x / strips;
  const int qs = blockIdx.x % strips;
  const int b  = bh >> 4, h = bh & 15;
  const int q0 = qs * QB + wv * QW;                 // wave's first query

  const bf16* qb  = q  + (size_t)bh * Nseq * Dh;
  const bf16* kbp = k  + (size_t)bh * Nseq * Dh;
  const bf16* vbp = vT + (size_t)bh * Dh * Nseq;

  // Q A-operands: 2 query tiles x (k0 = 0, 32)
  v16bf aq[2][2];
  #pragma unroll
  for (int qt = 0; qt < 2; ++qt) {
    aq[qt][0] = load_a(qb + (size_t)(q0 + qt * 16 + lr) * Dh + 0,  lane);
    aq[qt][1] = load_a(qb + (size_t)(q0 + qt * 16 + lr) * Dh + 32, lane);
  }

  v8f zv;
  float mrun[2][8], srun[2][8];
  v8f o[2][4];
  #pragma unroll
  for (int i = 0; i < 8; ++i) zv[i] = 0.f;
  #pragma unroll
  for (int qt = 0; qt < 2; ++qt) {
    #pragma unroll
    for (int i = 0; i < 8; ++i) { mrun[qt][i] = -1e30f; srun[qt][i] = 0.f; }
    #pragma unroll
    for (int t = 0; t < 4; ++t)
      #pragma unroll
      for (int i = 0; i < 8; ++i) o[qt][t][i] = 0.f;
  }

  for (int jb = 0; jb < Nseq; jb += KB) {
    // cooperative K/V -> LDS: 128 threads x 2 b128 chunks each per matrix
    #pragma unroll
    for (int c = 0; c < 2; ++c) {
      const int chunk = tid + c * 128;              // 0..255
      const int krow = chunk >> 3;                  // 8 chunks per 64-elem K row
      const int kcol = (chunk & 7) * 8;
      v8bf kv = *reinterpret_cast<const v8bf*>(kbp + (size_t)(jb + krow) * Dh + kcol);
      *reinterpret_cast<v8bf*>(&kls[krow * Dh + kcol]) = kv;
      const int vrow = chunk >> 2;                  // 4 chunks per 32-elem V row
      const int vcol = (chunk & 3) * 8;
      v8bf vv = *reinterpret_cast<const v8bf*>(vbp + (size_t)vrow * Nseq + jb + vcol);
      *reinterpret_cast<v8bf*>(&vls[vrow * KB + vcol]) = vv;
    }
    if (jb + KB < Nseq) {                           // hint next tiles toward L0/L2
      __builtin_prefetch(kbp + (size_t)(jb + KB + (tid >> 2)) * Dh, 0, 3);
      __builtin_prefetch(vbp + (size_t)(tid >> 1) * Nseq + jb + KB, 0, 3);
    }
    __syncthreads();

    // S = Q(16x64) * K^T(64x32): B operands from LDS, shared by all 4 waves
    v16bf kb00 = load_b(&kls[(lr)      * Dh + 0],  lane);
    v16bf kb01 = load_b(&kls[(lr)      * Dh + 32], lane);
    v16bf kb10 = load_b(&kls[(16 + lr) * Dh + 0],  lane);
    v16bf kb11 = load_b(&kls[(16 + lr) * Dh + 32], lane);

    #pragma unroll
    for (int qt = 0; qt < 2; ++qt) {
      v8f s0 = wmma_bf16(aq[qt][1], kb01, zv); s0 = wmma_bf16(aq[qt][0], kb00, s0);
      v8f s1 = wmma_bf16(aq[qt][1], kb11, zv); s1 = wmma_bf16(aq[qt][0], kb10, s1);

      #pragma unroll
      for (int i = 0; i < 8; ++i) {   // bf16 mask (stays hot in 192MB L2)
        const int mq = q0 + qt * 16 + i + half * 8;
        s0[i] += (float)maskb[(size_t)mq * Nseq + jb + lr];
        s1[i] += (float)maskb[(size_t)mq * Nseq + jb + 16 + lr];
      }

      #pragma unroll
      for (int i = 0; i < 8; ++i) {   // online softmax per query row
        float t = fmaxf(s0[i], s1[i]);
        t = fmaxf(t, __shfl_xor(t, 1, 32));
        t = fmaxf(t, __shfl_xor(t, 2, 32));
        t = fmaxf(t, __shfl_xor(t, 4, 32));
        t = fmaxf(t, __shfl_xor(t, 8, 32));
        const float mnew  = fmaxf(mrun[qt][i], t);
        const float alpha = __expf(mrun[qt][i] - mnew);
        mrun[qt][i] = mnew;
        const float p0 = __expf(s0[i] - mnew);
        const float p1 = __expf(s1[i] - mnew);
        float rs = p0 + p1;
        rs += __shfl_xor(rs, 1, 32);
        rs += __shfl_xor(rs, 2, 32);
        rs += __shfl_xor(rs, 4, 32);
        rs += __shfl_xor(rs, 8, 32);
        srun[qt][i] = srun[qt][i] * alpha + rs;
        #pragma unroll
        for (int t4 = 0; t4 < 4; ++t4) o[qt][t4][i] *= alpha;
        const int m = i + half * 8;
        pls[wv][qt * 512 + m * 32 + lr]      = (bf16)p0;  // P as [m][key]
        pls[wv][qt * 512 + m * 32 + 16 + lr] = (bf16)p1;
      }
    }
    __syncthreads();

    // O += P(16x32) * V(32keys x 16d): V B-operands shared, P from own LDS area
    v16bf bv[4];
    #pragma unroll
    for (int t = 0; t < 4; ++t)
      bv[t] = load_b(&vls[(t * 16 + lr) * KB], lane);
    #pragma unroll
    for (int qt = 0; qt < 2; ++qt) {
      BF16x16 up;
      const bf16* pp = &pls[wv][qt * 512 + lr * 32 + half * 8];
      up.h[0] = *reinterpret_cast<const v8bf*>(pp);
      up.h[1] = *reinterpret_cast<const v8bf*>(pp + 16);
      #pragma unroll
      for (int t = 0; t < 4; ++t)
        o[qt][t] = wmma_bf16(up.v, bv[t], o[qt][t]);
    }
    __syncthreads();   // K/V/P consumed before next tile overwrites
  }

  #pragma unroll
  for (int qt = 0; qt < 2; ++qt) {
    #pragma unroll
    for (int t = 0; t < 4; ++t) {
      #pragma unroll
      for (int i = 0; i < 8; ++i) {
        const int m  = i + half * 8;
        const int nq = q0 + qt * 16 + m;
        const int d  = t * 16 + lr;
        const float val = o[qt][t][i] / srun[qt][i];
        ao[((size_t)b * Nseq + nq) * Cch + h * Dh + d] = (bf16)val;
      }
    }
  }
}

// ---------------------------------------------------------------- out proj
__global__ void __launch_bounds__(256) proj_ker(
    const bf16* __restrict__ xb, const bf16* __restrict__ wb,
    const float* __restrict__ bias, float* __restrict__ out) {
  const int lane  = threadIdx.x & 31;
  const int lr    = lane & 15;
  const int half  = (lane >> 4) & 1;
  const int wid   = blockIdx.x * (blockDim.x >> 5) + (threadIdx.x >> 5);
  const int NT    = Cch / 64;                 // 16
  const int mbase = (wid / NT) * 32;
  const int nbase = (wid % NT) * 64;

  v8f c[2][4];
  #pragma unroll
  for (int a = 0; a < 2; ++a)
    #pragma unroll
    for (int t = 0; t < 4; ++t)
      #pragma unroll
      for (int i = 0; i < 8; ++i) c[a][t][i] = 0.f;

  for (int kk = 0; kk < Cch; kk += 32) {
    v16bf a0 = load_a(xb + (size_t)(mbase + lr) * Cch + kk, lane);
    v16bf a1 = load_a(xb + (size_t)(mbase + 16 + lr) * Cch + kk, lane);
    #pragma unroll
    for (int t = 0; t < 4; ++t) {
      v16bf bop = load_b(wb + (size_t)(nbase + t * 16 + lr) * Cch + kk, lane);
      c[0][t] = wmma_bf16(a0, bop, c[0][t]);
      c[1][t] = wmma_bf16(a1, bop, c[1][t]);
    }
  }

  #pragma unroll
  for (int a = 0; a < 2; ++a) {
    #pragma unroll
    for (int t = 0; t < 4; ++t) {
      #pragma unroll
      for (int i = 0; i < 8; ++i) {
        const int tok  = mbase + a * 16 + i + half * 8;
        const int ncol = nbase + t * 16 + lr;
        out[(size_t)tok * Cch + ncol] = c[a][t][i] + bias[ncol];
      }
    }
  }
}

// ---------------------------------------------------------------- launcher
extern "C" void kernel_launch(void* const* d_in, const int* in_sizes, int n_in,
                              void* d_out, int out_size, void* d_ws, size_t ws_size,
                              hipStream_t stream) {
  (void)in_sizes; (void)n_in; (void)out_size; (void)ws_size;
  const float* x     = (const float*)d_in[0];
  const float* fcos  = (const float*)d_in[1];
  const float* fsin  = (const float*)d_in[2];
  const float* mask  = (const float*)d_in[3];
  const float* Wqkv  = (const float*)d_in[4];
  const float* Wproj = (const float*)d_in[5];
  const float* bproj = (const float*)d_in[6];
  float* out = (float*)d_out;

  char* ws = (char*)d_ws;
  size_t off = 0;
  auto carve = [&](size_t bytes) -> void* {
    void* p = ws + off;
    off += (bytes + 255) & ~(size_t)255;
    return p;
  };
  bf16* xb     = (bf16*)carve((size_t)ROWS * Cch * 2);          // 16 MB
  bf16* wqkvb  = (bf16*)carve((size_t)3 * Cch * Cch * 2);       //  6 MB
  bf16* wprojb = (bf16*)carve((size_t)Cch * Cch * 2);           //  2 MB
  bf16* maskb  = (bf16*)carve((size_t)Nseq * Nseq * 2);         //  8 MB
  bf16* qbuf   = (bf16*)carve((size_t)Bn * Hh * Nseq * Dh * 2); // 16 MB
  bf16* kbuf   = (bf16*)carve((size_t)Bn * Hh * Nseq * Dh * 2); // 16 MB
  bf16* vTbuf  = (bf16*)carve((size_t)Bn * Hh * Dh * Nseq * 2); // 16 MB
  bf16* aobuf  = (bf16*)carve((size_t)ROWS * Cch * 2);          // 16 MB

  // 1) f32 -> bf16 staging (x, weights, mask)
  {
    int n = ROWS * Cch;
    cvt_ker<<<(n / 4 + 255) / 256, 256, 0, stream>>>(x, xb, n);
    n = 3 * Cch * Cch;
    cvt_ker<<<(n / 4 + 255) / 256, 256, 0, stream>>>(Wqkv, wqkvb, n);
    n = Cch * Cch;
    cvt_ker<<<(n / 4 + 255) / 256, 256, 0, stream>>>(Wproj, wprojb, n);
    n = Nseq * Nseq;
    cvt_ker<<<(n / 4 + 255) / 256, 256, 0, stream>>>(mask, maskb, n);
  }

  // 2) QKV GEMM + RoPE (waves = (8192/32) * (3072/64) = 12288; 8 waves/block)
  qkv_rope_ker<<<(ROWS / 32) * ((3 * Cch) / 64) / 8, 256, 0, stream>>>(
      xb, wqkvb, fcos, fsin, qbuf, kbuf, vTbuf);

  // 3) Flash attention: (B*H) * (N/128) = 1024 blocks of 4 waves
  attn_ker<<<(Bn * Hh) * (Nseq / QB), 128, 0, stream>>>(
      qbuf, kbuf, vTbuf, maskb, aobuf);

  // 4) Output projection (waves = (8192/32) * (1024/64) = 4096; 8 waves/block)
  proj_ker<<<(ROWS / 32) * (Cch / 64) / 8, 256, 0, stream>>>(
      aobuf, wprojb, bproj, out);
}
